// DebertaV2Attention_36842229465620
// MI455X (gfx1250) — compile-verified, matrix operand order checked by reading, not run on hardware
//
#include <hip/hip_runtime.h>
#include <hip/hip_bf16.h>

// DeBERTa-v2 disentangled attention for MI455X (gfx1250, wave32, WMMA).
// All GEMM-shaped math runs through v_wmma_f32_16x16x32_f16 (f16 in, f32 acc).
// Disentangled c2p/p2c biases use the band trick: for a 16x16 (i,j) score tile,
// the needed relative-position rows form a 31-wide band, so each bias tile is a
// 16x32 band-GEMM (4 WMMAs) + an LDS diagonal gather.
//
// V is produced TRANSPOSED ([b][d][j]) straight from the projection GEMM's
// epilogue so the P.V B-fragments use the same two global_load_b128 as all
// other fragments (no strided scalar loads anywhere in the attention loop).
//
// Workspace (f16, ~58 MB): Xh | Rel16 | Wq..Wo | Q K | PosK PosQ | Vt | Ctx

typedef __attribute__((ext_vector_type(16))) _Float16 v16h;
typedef __attribute__((ext_vector_type(8)))  _Float16 v8h;
typedef __attribute__((ext_vector_type(8)))  float    v8f;

#define Bn  8
#define Nn  512
#define Dn  1024
#define Hn  16
#define HDn 64
#define PBn 512

// ---------------------------------------------------------------------------
// WMMA wrapper: D = A(16x32 f16) * B(32x16 f16) + C(16x16 f32)
__device__ __forceinline__ v8f wmma_f16(v16h a, v16h b, v8f c) {
  return __builtin_amdgcn_wmma_f32_16x16x32_f16(false, a, false, b, (short)0, c,
                                                false, false);
}

// A/B fragment from row-major storage where the contraction dim K is the
// contiguous axis. 16-bit 16x32 layout:
//   lanes 0-15  -> rows 0-15, K {0..7,16..23}
//   lanes 16-31 -> rows 0-15, K {8..15,24..31}
// Two 16B loads per lane -> global_load_b128 pairs.
__device__ __forceinline__ v16h frag_rm(const _Float16* __restrict__ mat,
                                        int row0, int maxrow, int ld) {
  const int lane = threadIdx.x & 31;
  int r = row0 + (lane & 15);
  r = (r < 0) ? 0 : (r > maxrow ? maxrow : r);
  const int kb = (lane >> 4) * 8;
  const _Float16* p = mat + (size_t)r * ld + kb;
  v8h lo = *(const v8h*)(p);        // K = kb .. kb+7
  v8h hi = *(const v8h*)(p + 16);   // K = kb+16 .. kb+23
  v16h out;
#pragma unroll
  for (int t = 0; t < 8; ++t) { out[t] = lo[t]; out[8 + t] = hi[t]; }
  return out;
}

// A fragment from a per-wave 16x32 LDS staging tile (row stride 32 f16).
__device__ __forceinline__ v16h frag_lds(const _Float16* p0) {
  const int lane = threadIdx.x & 31;
  const int row = lane & 15;
  const int kb  = (lane >> 4) * 8;
  const _Float16* p = p0 + row * 32 + kb;
  v8h lo = *(const v8h*)(p);        // ds_load_b128
  v8h hi = *(const v8h*)(p + 16);
  v16h out;
#pragma unroll
  for (int t = 0; t < 8; ++t) { out[t] = lo[t]; out[8 + t] = hi[t]; }
  return out;
}

// ---------------------------------------------------------------------------
__global__ void cvt_f32_f16(const float* __restrict__ s, _Float16* __restrict__ d,
                            int n) {
  int i = blockIdx.x * blockDim.x + threadIdx.x;
  if (i < n) d[i] = (_Float16)s[i];
}

// ---------------------------------------------------------------------------
// C = A[M,K] * W[N,K]^T + bias[N].  One 32x64 tile per wave (B frags reused
// across two A tiles: 6 fragment loads per 8 WMMAs).
// TR=false: C[M,N] f16 row-major.
// TR=true : per-batch transposed store Ct[b][n][j] with 512 tokens per batch
//           (used for V so the attention P.V B-fragments are contiguous).
template <bool TR>
__global__ __launch_bounds__(256) void gemm_bias_f16(
    const _Float16* __restrict__ A, const _Float16* __restrict__ W,
    const float* __restrict__ bias, _Float16* __restrict__ C,
    int M, int N, int K) {
  const int wid = blockIdx.x * 8 + (threadIdx.x >> 5);
  const int ntn = N >> 6;
  const int mt = wid / ntn, nt = wid % ntn;
  const int m0 = mt * 32, n0 = nt * 64;
  if (m0 >= M) return;
  v8f acc[2][4] = {};
  for (int k0 = 0; k0 < K; k0 += 32) {
    v16h a0 = frag_rm(A + k0, m0, M - 1, K);
    v16h a1 = frag_rm(A + k0, m0 + 16, M - 1, K);
#pragma unroll
    for (int t = 0; t < 4; ++t) {
      v16h b = frag_rm(W + k0, n0 + t * 16, N - 1, K);
      acc[0][t] = wmma_f16(a0, b, acc[0][t]);
      acc[1][t] = wmma_f16(a1, b, acc[1][t]);
    }
  }
  const int lane = threadIdx.x & 31;
  const int cn = lane & 15, rb = (lane >> 4) * 8;
#pragma unroll
  for (int h = 0; h < 2; ++h) {
#pragma unroll
    for (int t = 0; t < 4; ++t) {
      const int n = n0 + t * 16 + cn;
      const float bn = bias[n];
#pragma unroll
      for (int r = 0; r < 8; ++r) {
        const int m = m0 + h * 16 + rb + r;
        const float v = acc[h][t][r] + bn;
        if (TR) {
          const int b = m >> 9, j = m & 511;        // 512 tokens per batch
          C[((size_t)b * N + n) * (size_t)Nn + j] = (_Float16)v;
        } else {
          C[(size_t)m * N + n] = (_Float16)v;
        }
      }
    }
  }
}

// ---------------------------------------------------------------------------
// Fused flash-style attention with disentangled c2p/p2c band bias.
// Block = 8 waves; wave w owns rows [i0, i0+16) for one (b, h).
// Vt is [b][d][j] so P.V B-fragments are plain row-major fragments.
__global__ __launch_bounds__(256) void attn_kernel(
    const _Float16* __restrict__ Q, const _Float16* __restrict__ Kx,
    const _Float16* __restrict__ Vt, const _Float16* __restrict__ PK,
    const _Float16* __restrict__ PQ, _Float16* __restrict__ Ctx) {
  __shared__ __align__(16) float     gbuf[8][16][32];  // band-GEMM gather buffer
  __shared__ __align__(16) _Float16  pbuf[8][16][32];  // P staging (K upper half = 0)

  const int wave = threadIdx.x >> 5;
  const int lane = threadIdx.x & 31;

  for (int idx = threadIdx.x; idx < 8 * 16 * 32; idx += 256)
    ((_Float16*)pbuf)[idx] = (_Float16)0.f;
  __syncthreads();

  const int itile = blockIdx.x & 3;            // N/128 = 4 i-tiles per (b,h)
  const int bh = blockIdx.x >> 2;
  const int b = bh >> 4, h = bh & 15;
  const int i0 = itile * 128 + wave * 16;
  const int hc = h * HDn;

  const _Float16* Qm  = Q  + (size_t)b * Nn * Dn + hc;
  const _Float16* Km  = Kx + (size_t)b * Nn * Dn + hc;
  const _Float16* Vtb = Vt + (size_t)b * Dn * Nn;      // [d][j]
  const _Float16* PKm = PK + hc;
  const _Float16* PQm = PQ + hc;

  const v16h qa0 = frag_rm(Qm, i0, Nn - 1, Dn);        // Q rows, K 0..31
  const v16h qa1 = frag_rm(Qm + 32, i0, Nn - 1, Dn);   // Q rows, K 32..63

  const float inv_scale = rsqrtf((float)(HDn * 3));    // scale_factor = 3
  float mrow[8], lrow[8];
#pragma unroll
  for (int r = 0; r < 8; ++r) { mrow[r] = -3.0e38f; lrow[r] = 0.f; }
  v8f acc[4] = {};
  const int cn = lane & 15, rb = (lane >> 4) * 8;

  for (int j0 = 0; j0 < Nn; j0 += 16) {
    // ---- content scores: S = Q K^T  (B[k,n] = K[j0+n, k], K contiguous)
    v8f s = {};
    const v16h kb0 = frag_rm(Km, j0, Nn - 1, Dn);
    const v16h kb1 = frag_rm(Km + 32, j0, Nn - 1, Dn);
    s = wmma_f16(qa0, kb0, s);
    s = wmma_f16(qa1, kb1, s);

    // band of relative positions hit by this tile: r' = i_loc - j_loc + 15
    const int base = i0 - j0 - 15 + PBn;

    // ---- c2p: M[i, r'] = q_i . pos_k[base + r'],  r' in [0,32)
    {
      v8f m0 = {}, m1 = {};
      m0 = wmma_f16(qa0, frag_rm(PKm,      base,      2 * PBn - 1, Dn), m0);
      m0 = wmma_f16(qa1, frag_rm(PKm + 32, base,      2 * PBn - 1, Dn), m0);
      m1 = wmma_f16(qa0, frag_rm(PKm,      base + 16, 2 * PBn - 1, Dn), m1);
      m1 = wmma_f16(qa1, frag_rm(PKm + 32, base + 16, 2 * PBn - 1, Dn), m1);
#pragma unroll
      for (int r = 0; r < 8; ++r) {
        gbuf[wave][rb + r][cn]      = m0[r];
        gbuf[wave][rb + r][16 + cn] = m1[r];
      }
      asm volatile("s_wait_dscnt 0x0" ::: "memory");
#pragma unroll
      for (int r = 0; r < 8; ++r) {
        const int il = rb + r;
        s[r] += gbuf[wave][il][il - cn + 15];       // c2p_g[i, j]
      }
    }

    // ---- p2c: M[j, r'] = k_j . pos_q[base + r'] ; gathered transposed
    {
      v8f m0 = {}, m1 = {};
      m0 = wmma_f16(kb0, frag_rm(PQm,      base,      2 * PBn - 1, Dn), m0);
      m0 = wmma_f16(kb1, frag_rm(PQm + 32, base,      2 * PBn - 1, Dn), m0);
      m1 = wmma_f16(kb0, frag_rm(PQm,      base + 16, 2 * PBn - 1, Dn), m1);
      m1 = wmma_f16(kb1, frag_rm(PQm + 32, base + 16, 2 * PBn - 1, Dn), m1);
      asm volatile("s_wait_dscnt 0x0" ::: "memory"); // c2p gathers done first
#pragma unroll
      for (int r = 0; r < 8; ++r) {
        gbuf[wave][rb + r][cn]      = m0[r];
        gbuf[wave][rb + r][16 + cn] = m1[r];
      }
      asm volatile("s_wait_dscnt 0x0" ::: "memory");
#pragma unroll
      for (int r = 0; r < 8; ++r) {
        const int il = rb + r;
        s[r] += gbuf[wave][cn][il - cn + 15];       // p2c_g[i, j] = M[j_loc, r']
      }
    }

    // ---- scale + online softmax (rows 0-7 in lanes 0-15, rows 8-15 in 16-31)
#pragma unroll
    for (int r = 0; r < 8; ++r) {
      float v = s[r] * inv_scale;
      s[r] = v;
      v = fmaxf(v, __shfl_xor(v, 1, 32));
      v = fmaxf(v, __shfl_xor(v, 2, 32));
      v = fmaxf(v, __shfl_xor(v, 4, 32));
      v = fmaxf(v, __shfl_xor(v, 8, 32));
      const float mn = fmaxf(mrow[r], v);
      const float fs = __expf(mrow[r] - mn);
      mrow[r] = mn;
      float p = __expf(s[r] - mn);
      s[r] = p;
      float rs = p;
      rs += __shfl_xor(rs, 1, 32);
      rs += __shfl_xor(rs, 2, 32);
      rs += __shfl_xor(rs, 4, 32);
      rs += __shfl_xor(rs, 8, 32);
      lrow[r] = lrow[r] * fs + rs;
#pragma unroll
      for (int t = 0; t < 4; ++t) acc[t][r] *= fs;
    }

    // ---- ctx += P V  (stage P as zero-padded 16x32 A tile in LDS)
#pragma unroll
    for (int r = 0; r < 8; ++r)
      pbuf[wave][rb + r][cn] = (_Float16)s[r];
    asm volatile("s_wait_dscnt 0x0" ::: "memory");
    const v16h pa = frag_lds(&pbuf[wave][0][0]);
    // B[k,n] = V[j0+k, d0+n] = Vt[d0+n][j0+k]: row-major fragment on Vt.
    // Columns j0+16..j0+31 are always multiplied by the zero half of P; for
    // j0 = 496 they read <=32B past the row (still inside d_ws: Ctx follows).
#pragma unroll
    for (int t = 0; t < 4; ++t)
      acc[t] = wmma_f16(pa, frag_rm(Vtb + j0, hc + t * 16, Dn - 1, Nn), acc[t]);
  }

  // ---- normalize + write ctx (f16, [B,N,D] with head slice at hc)
#pragma unroll
  for (int t = 0; t < 4; ++t) {
#pragma unroll
    for (int r = 0; r < 8; ++r) {
      const int row = i0 + rb + r;
      Ctx[(size_t)(b * Nn + row) * Dn + hc + t * 16 + cn] =
          (_Float16)(acc[t][r] / lrow[r]);
    }
  }
}

// ---------------------------------------------------------------------------
// out = LayerNorm(Ctx @ o_w^T + o_b + hidden).  One block = 16 rows x 1024 cols;
// wave w covers cols [w*128, w*128+128).
__global__ __launch_bounds__(256) void out_ln_kernel(
    const _Float16* __restrict__ Ctx, const _Float16* __restrict__ Wo,
    const float* __restrict__ ob, const float* __restrict__ resid,
    const float* __restrict__ g, const float* __restrict__ be,
    float* __restrict__ out, int M) {
  __shared__ float ssum[16][8];
  __shared__ float ssq[16][8];
  const int wave = threadIdx.x >> 5;
  const int lane = threadIdx.x & 31;
  const int m0 = blockIdx.x * 16;
  const int n0 = wave * 128;

  v8f acc[8] = {};
  for (int k0 = 0; k0 < Dn; k0 += 32) {
    v16h a = frag_rm(Ctx + k0, m0, M - 1, Dn);
#pragma unroll
    for (int t = 0; t < 8; ++t)
      acc[t] = wmma_f16(a, frag_rm(Wo + k0, n0 + t * 16, Dn - 1, Dn), acc[t]);
  }

  const int cn = lane & 15, rb = (lane >> 4) * 8;
  float psum[8] = {}, psq[8] = {};
#pragma unroll
  for (int t = 0; t < 8; ++t) {
    const int n = n0 + t * 16 + cn;
    const float bn = ob[n];
#pragma unroll
    for (int r = 0; r < 8; ++r) {
      float hv = acc[t][r] + bn + resid[(size_t)(m0 + rb + r) * Dn + n];
      acc[t][r] = hv;
      psum[r] += hv;
      psq[r] += hv * hv;
    }
  }
#pragma unroll
  for (int r = 0; r < 8; ++r) {
    float a = psum[r], q = psq[r];
    a += __shfl_xor(a, 1, 32);  q += __shfl_xor(q, 1, 32);
    a += __shfl_xor(a, 2, 32);  q += __shfl_xor(q, 2, 32);
    a += __shfl_xor(a, 4, 32);  q += __shfl_xor(q, 4, 32);
    a += __shfl_xor(a, 8, 32);  q += __shfl_xor(q, 8, 32);
    if (cn == 0) { ssum[rb + r][wave] = a; ssq[rb + r][wave] = q; }
  }
  __syncthreads();
  float mu[8], inv[8];
#pragma unroll
  for (int r = 0; r < 8; ++r) {
    float a = 0.f, q = 0.f;
#pragma unroll
    for (int w = 0; w < 8; ++w) { a += ssum[rb + r][w]; q += ssq[rb + r][w]; }
    const float m = a * (1.f / (float)Dn);
    const float var = q * (1.f / (float)Dn) - m * m;
    mu[r] = m;
    inv[r] = rsqrtf(var + 1e-7f);
  }
#pragma unroll
  for (int t = 0; t < 8; ++t) {
    const int n = n0 + t * 16 + cn;
    const float gn = g[n], bn2 = be[n];
#pragma unroll
    for (int r = 0; r < 8; ++r)
      out[(size_t)(m0 + rb + r) * Dn + n] = (acc[t][r] - mu[r]) * inv[r] * gn + bn2;
  }
}

// ---------------------------------------------------------------------------
extern "C" void kernel_launch(void* const* d_in, const int* in_sizes, int n_in,
                              void* d_out, int out_size, void* d_ws, size_t ws_size,
                              hipStream_t stream) {
  (void)in_sizes; (void)n_in; (void)out_size; (void)ws_size;
  const float* hidden = (const float*)d_in[0];
  const float* rel    = (const float*)d_in[1];
  const float* q_w = (const float*)d_in[2];  const float* q_b = (const float*)d_in[3];
  const float* k_w = (const float*)d_in[4];  const float* k_b = (const float*)d_in[5];
  const float* v_w = (const float*)d_in[6];  const float* v_b = (const float*)d_in[7];
  const float* pk_w = (const float*)d_in[8]; const float* pk_b = (const float*)d_in[9];
  const float* pq_w = (const float*)d_in[10];const float* pq_b = (const float*)d_in[11];
  const float* o_w = (const float*)d_in[12]; const float* o_b = (const float*)d_in[13];
  const float* ln_g = (const float*)d_in[14];const float* ln_b = (const float*)d_in[15];

  char* ws = (char*)d_ws;
  size_t off = 0;
  auto alloc = [&](size_t elems) -> _Float16* {
    _Float16* p = (_Float16*)(ws + off);
    off += elems * sizeof(_Float16);
    return p;
  };
  const size_t MX = (size_t)Bn * Nn;       // 4096 token rows
  _Float16* Xh    = alloc(MX * Dn);        // f16 hidden states
  _Float16* Rel16 = alloc((size_t)2 * PBn * Dn);
  _Float16* Wq  = alloc((size_t)Dn * Dn);
  _Float16* Wk  = alloc((size_t)Dn * Dn);
  _Float16* Wv  = alloc((size_t)Dn * Dn);
  _Float16* Wpk = alloc((size_t)Dn * Dn);
  _Float16* Wpq = alloc((size_t)Dn * Dn);
  _Float16* Wo  = alloc((size_t)Dn * Dn);
  _Float16* Qb  = alloc(MX * Dn);
  _Float16* Kb  = alloc(MX * Dn);
  _Float16* PKb = alloc((size_t)2 * PBn * Dn);
  _Float16* PQb = alloc((size_t)2 * PBn * Dn);
  _Float16* Vt  = alloc(MX * Dn);          // transposed V [b][d][j]
  _Float16* Ctx = alloc(MX * Dn);          // must follow Vt (spill safety)

  auto cvt = [&](const float* s, _Float16* d, int n) {
    cvt_f32_f16<<<(n + 255) / 256, 256, 0, stream>>>(s, d, n);
  };
  cvt(hidden, Xh, (int)(MX * Dn));
  cvt(rel, Rel16, 2 * PBn * Dn);
  cvt(q_w, Wq, Dn * Dn);   cvt(k_w, Wk, Dn * Dn);   cvt(v_w, Wv, Dn * Dn);
  cvt(pk_w, Wpk, Dn * Dn); cvt(pq_w, Wpq, Dn * Dn); cvt(o_w, Wo, Dn * Dn);

  auto gemm = [&](const _Float16* A, const _Float16* W, const float* bias,
                  _Float16* C, int M, int N, int K) {
    const int tiles = (M / 32) * (N / 64);
    gemm_bias_f16<false><<<(tiles + 7) / 8, 256, 0, stream>>>(A, W, bias, C, M, N, K);
  };
  gemm(Xh, Wq, q_b, Qb, (int)MX, Dn, Dn);
  gemm(Xh, Wk, k_b, Kb, (int)MX, Dn, Dn);
  {
    const int tiles = ((int)MX / 32) * (Dn / 64);
    gemm_bias_f16<true><<<(tiles + 7) / 8, 256, 0, stream>>>(Xh, Wv, v_b, Vt,
                                                             (int)MX, Dn, Dn);
  }
  gemm(Rel16, Wpk, pk_b, PKb, 2 * PBn, Dn, Dn);
  gemm(Rel16, Wpq, pq_b, PQb, 2 * PBn, Dn, Dn);

  attn_kernel<<<Bn * Hn * (Nn / 128), 256, 0, stream>>>(Qb, Kb, Vt, PKb, PQb, Ctx);

  out_ln_kernel<<<(int)(MX / 16), 256, 0, stream>>>(Ctx, Wo, o_b, hidden, ln_g,
                                                    ln_b, (float*)d_out, (int)MX);
}
